// Generator_22806276341768
// MI455X (gfx1250) — compile-verified, matrix operand order checked by reading, not run on hardware
//
#include <hip/hip_runtime.h>
#include <hip/hip_bf16.h>

// ---------------------------------------------------------------------------
// Shapes from the reference
// ---------------------------------------------------------------------------
#define BATCH   256
#define TQ      128
#define NWORD   300
#define NH      512
#define HDIM    256      // H = NH/2
#define G4      1024     // 4*H gate width
#define MC_     64
#define LC_     8
#define MT_     16
#define LT_     6

// ---------------------------------------------------------------------------
// CDNA5 WMMA types
// ---------------------------------------------------------------------------
typedef __attribute__((ext_vector_type(16))) __bf16 v16bf;
typedef __attribute__((ext_vector_type(8)))  __bf16 v8bf;
typedef __attribute__((ext_vector_type(8)))  float  v8f;

__device__ inline __bf16 f2bf(float f) {
    unsigned u = __builtin_bit_cast(unsigned, f);
    u += 0x7FFFu + ((u >> 16) & 1u);          // round-to-nearest-even
    unsigned short hs = (unsigned short)(u >> 16);
    return __builtin_bit_cast(__bf16, hs);
}

__device__ inline v8bf pack8(const float* v) {
    v8bf r;
    #pragma unroll
    for (int i = 0; i < 8; ++i) r[i] = f2bf(v[i]);
    return r;
}

// A/B 16-bit fragment layout: lanes 0-15 hold K{0..7,16..23},
// lanes 16-31 hold K{8..15,24..31} of row (lane&15).
__device__ inline v16bf load_frag(const __bf16* base, int laneHalf) {
    const v8bf lo = *(const v8bf*)(base + laneHalf * 8);
    const v8bf hi = *(const v8bf*)(base + 16 + laneHalf * 8);
    v16bf r;
    #pragma unroll
    for (int i = 0; i < 8; ++i) { r[i] = lo[i]; r[8 + i] = hi[i]; }
    return r;
}

// ---------------------------------------------------------------------------
// Generic batched GEMM:  C[z] (=|+=) A[z][M,K] * B[z][N,K]^T (+ bias[N])
// fp32 in/out, bf16 WMMA (v_wmma_f32_16x16x32_bf16) with f32 accumulate.
// Tile 64x64, 4 waves (wave32), K-step 32, LDS-staged bf16 tiles.
// Contract: A/B base pointers 16B-aligned, lda/ldb multiples of 4 floats.
// Edge rows/cols stage garbage from a clamped in-bounds row (harmless: those
// accumulator rows/cols are never stored); only the K tail is zero-filled.
// ---------------------------------------------------------------------------
#define TM 64
#define TN 64
#define TK 32
#define LDSP 40   // row stride in bf16 elems: 80B rows keep 16B chunk alignment

// branch-free: 16 floats -> 16 bf16 in LDS
__device__ inline void stage16_fast(__bf16* dst, const float* src) {
    const float4* p = (const float4*)src;
    const float4 q0 = p[0], q1 = p[1], q2 = p[2], q3 = p[3];
    float buf[16];
    buf[0]=q0.x;  buf[1]=q0.y;  buf[2]=q0.z;  buf[3]=q0.w;
    buf[4]=q1.x;  buf[5]=q1.y;  buf[6]=q1.z;  buf[7]=q1.w;
    buf[8]=q2.x;  buf[9]=q2.y;  buf[10]=q2.z; buf[11]=q2.w;
    buf[12]=q3.x; buf[13]=q3.y; buf[14]=q3.z; buf[15]=q3.w;
    *(v8bf*)(dst)     = pack8(&buf[0]);
    *(v8bf*)(dst + 8) = pack8(&buf[8]);
}

// K-tail: zero beyond K
__device__ inline void stage16_tail(__bf16* dst, const float* src, int kBase, int K) {
    float buf[16];
    #pragma unroll
    for (int c = 0; c < 16; ++c)
        buf[c] = (kBase + c < K) ? src[c] : 0.f;
    *(v8bf*)(dst)     = pack8(&buf[0]);
    *(v8bf*)(dst + 8) = pack8(&buf[8]);
}

__global__ __launch_bounds__(128)
void gemm_bt_kernel(const float* __restrict__ A, const float* __restrict__ Bw,
                    float* __restrict__ C, const float* __restrict__ bias,
                    int M, int N, int K, int lda, int ldb, int ldc,
                    long strideA, long strideB, long strideC, int accum)
{
    __shared__ __align__(16) __bf16 As[TM][LDSP];
    __shared__ __align__(16) __bf16 Bs[TN][LDSP];

    const int tid      = threadIdx.x;
    const int wave     = tid >> 5;
    const int lane     = tid & 31;
    const int laneRow  = lane & 15;
    const int laneHalf = lane >> 4;

    const int z = blockIdx.z;
    A  += (long)z * strideA;
    Bw += (long)z * strideB;
    C  += (long)z * strideC;

    const int m0    = blockIdx.y * TM;
    const int n0    = blockIdx.x * TN;
    const int waveM = m0 + wave * 16;

    v8f acc[4];
    #pragma unroll
    for (int j = 0; j < 4; ++j) {
        const int col     = n0 + j * 16 + laneRow;
        const int rowBase = waveM + laneHalf * 8;
        if (accum) {
            #pragma unroll
            for (int r = 0; r < 8; ++r) {
                const int row = rowBase + r;
                acc[j][r] = (row < M && col < N) ? C[(long)row * ldc + col] : 0.f;
            }
        } else {
            const float bv = (bias && col < N) ? bias[col] : 0.f;
            #pragma unroll
            for (int r = 0; r < 8; ++r) acc[j][r] = bv;
        }
    }

    const int stRow = tid >> 1;          // 0..63
    const int stCol = (tid & 1) * 16;    // 0 or 16
    const int aRow  = m0 + stRow;  const int aRowC = (aRow < M) ? aRow : (M - 1);
    const int bRow  = n0 + stRow;  const int bRowC = (bRow < N) ? bRow : (N - 1);
    const float* aSrcRow = A  + (long)aRowC * lda + stCol;
    const float* bSrcRow = Bw + (long)bRowC * ldb + stCol;

    __bf16* aDst = &As[stRow][stCol];
    __bf16* bDst = &Bs[stRow][stCol];

    const int kFull = K & ~(TK - 1);

    for (int k0 = 0; k0 < kFull; k0 += TK) {
        // speculative prefetch of next K tile (OOB prefetches silently dropped)
        __builtin_prefetch(aSrcRow + k0 + TK, 0, 1);
        __builtin_prefetch(bSrcRow + k0 + TK, 0, 1);
        __syncthreads();
        stage16_fast(aDst, aSrcRow + k0);
        stage16_fast(bDst, bSrcRow + k0);
        __syncthreads();

        const v16bf af  = load_frag(&As[wave * 16 + laneRow][0], laneHalf);
        const v16bf bf0 = load_frag(&Bs[ 0 + laneRow][0], laneHalf);
        const v16bf bf1 = load_frag(&Bs[16 + laneRow][0], laneHalf);
        const v16bf bf2 = load_frag(&Bs[32 + laneRow][0], laneHalf);
        const v16bf bf3 = load_frag(&Bs[48 + laneRow][0], laneHalf);
        acc[0] = __builtin_amdgcn_wmma_f32_16x16x32_bf16(
            false, af, false, bf0, (short)0, acc[0], false, false);
        acc[1] = __builtin_amdgcn_wmma_f32_16x16x32_bf16(
            false, af, false, bf1, (short)0, acc[1], false, false);
        acc[2] = __builtin_amdgcn_wmma_f32_16x16x32_bf16(
            false, af, false, bf2, (short)0, acc[2], false, false);
        acc[3] = __builtin_amdgcn_wmma_f32_16x16x32_bf16(
            false, af, false, bf3, (short)0, acc[3], false, false);
    }

    if (kFull < K) {   // peeled K tail (only K % 32 != 0, e.g. K = 300)
        __syncthreads();
        stage16_tail(aDst, aSrcRow + kFull, kFull + stCol, K);
        stage16_tail(bDst, bSrcRow + kFull, kFull + stCol, K);
        __syncthreads();

        const v16bf af  = load_frag(&As[wave * 16 + laneRow][0], laneHalf);
        const v16bf bf0 = load_frag(&Bs[ 0 + laneRow][0], laneHalf);
        const v16bf bf1 = load_frag(&Bs[16 + laneRow][0], laneHalf);
        const v16bf bf2 = load_frag(&Bs[32 + laneRow][0], laneHalf);
        const v16bf bf3 = load_frag(&Bs[48 + laneRow][0], laneHalf);
        acc[0] = __builtin_amdgcn_wmma_f32_16x16x32_bf16(
            false, af, false, bf0, (short)0, acc[0], false, false);
        acc[1] = __builtin_amdgcn_wmma_f32_16x16x32_bf16(
            false, af, false, bf1, (short)0, acc[1], false, false);
        acc[2] = __builtin_amdgcn_wmma_f32_16x16x32_bf16(
            false, af, false, bf2, (short)0, acc[2], false, false);
        acc[3] = __builtin_amdgcn_wmma_f32_16x16x32_bf16(
            false, af, false, bf3, (short)0, acc[3], false, false);
    }

    #pragma unroll
    for (int j = 0; j < 4; ++j) {
        const int col = n0 + j * 16 + laneRow;
        if (col >= N) continue;
        const int rowBase = waveM + laneHalf * 8;
        #pragma unroll
        for (int r = 0; r < 8; ++r) {
            const int row = rowBase + r;
            if (row < M) C[(long)row * ldc + col] = acc[j][r];
        }
    }
}

// ---------------------------------------------------------------------------
// Elementwise / data-movement kernels
// ---------------------------------------------------------------------------
__global__ void fill_kernel(float* p, long n, float v) {
    long id = (long)blockIdx.x * blockDim.x + threadIdx.x;
    if (id < n) p[id] = v;
}

// dst[n,t,:] = src[n, rev? (t<len? len-1-t : t) : t, :], arbitrary strides
__global__ void seq_copy_kernel(float* __restrict__ dst, const float* __restrict__ src,
                                const int* __restrict__ lens,
                                int Nb, int T, int D,
                                long sN, long sT, long dN, long dT, int rev)
{
    long id = (long)blockIdx.x * blockDim.x + threadIdx.x;
    long total = (long)Nb * T * D;
    if (id >= total) return;
    const int  d = (int)(id % D);
    const long r = id / D;
    const int  t = (int)(r % T);
    const int  n = (int)(r / T);
    int ts = t;
    if (rev) { const int L = lens[n]; ts = (t < L) ? (L - 1 - t) : t; }
    dst[(long)n * dN + (long)t * dT + d] = src[(long)n * sN + (long)ts * sT + d];
}

// packed-sequence LSTM cell: g already holds x@Wih^T + h@Whh^T
__global__ void lstm_gate_kernel(const float* __restrict__ g,
                                 const float* __restrict__ b_ih,
                                 const float* __restrict__ b_hh,
                                 float* __restrict__ h, float* __restrict__ c,
                                 const int* __restrict__ lens, int t, int Nb,
                                 float* __restrict__ y, long yN, long yT)
{
    long id = (long)blockIdx.x * blockDim.x + threadIdx.x;
    if (id >= (long)Nb * HDIM) return;
    const int j = (int)(id % HDIM);
    const int n = (int)(id / HDIM);
    const float* gr = g + (long)n * G4;
    const float gi = gr[j]            + b_ih[j]            + b_hh[j];
    const float gf = gr[HDIM + j]     + b_ih[HDIM + j]     + b_hh[HDIM + j];
    const float gg = gr[2 * HDIM + j] + b_ih[2 * HDIM + j] + b_hh[2 * HDIM + j];
    const float go = gr[3 * HDIM + j] + b_ih[3 * HDIM + j] + b_hh[3 * HDIM + j];
    const float si = 1.f / (1.f + expf(-gi));
    const float sf = 1.f / (1.f + expf(-gf));
    const float so = 1.f / (1.f + expf(-go));
    const float cn = sf * c[id] + si * tanhf(gg);
    const float hn = so * tanhf(cn);
    float* yp = y + (long)n * yN + (long)t * yT + j;
    if (t < lens[n]) { c[id] = cn; h[id] = hn; *yp = hn; }
    else             { *yp = 0.f; }                        // carry frozen, output 0
}

__global__ void take_last_kernel(float* __restrict__ dst, const float* __restrict__ src,
                                 const int* __restrict__ lens, int Nb, int D,
                                 long sN, long sT)
{
    long id = (long)blockIdx.x * blockDim.x + threadIdx.x;
    if (id >= (long)Nb * D) return;
    const int d = (int)(id % D);
    const int n = (int)(id / D);
    const int t = lens[n] - 1;
    dst[id] = src[(long)n * sN + (long)t * sT + d];
}

// col_enc[b,c,:] += tab_enc[b, col2tab[b,c], :]
__global__ void add_parent_kernel(float* __restrict__ col_enc,
                                  const float* __restrict__ tab_enc,
                                  const int* __restrict__ col2tab, int D)
{
    long id = (long)blockIdx.x * blockDim.x + threadIdx.x;
    if (id >= (long)BATCH * MC_ * D) return;
    const int  d = (int)(id % D);
    const long r = id / D;
    const int  ci = (int)(r % MC_);
    const int  b  = (int)(r / MC_);
    const int  ti = col2tab[(long)b * MC_ + ci];
    col_enc[id] += tab_enc[((long)b * MT_ + ti) * D + d];
}

// ---------------------------------------------------------------------------
// Host-side orchestration
// ---------------------------------------------------------------------------
struct Bump { char* base; size_t off; };
static inline float* balloc(Bump& b, size_t nFloats) {
    float* p = (float*)(b.base + b.off);
    b.off += ((nFloats * sizeof(float)) + 255) & ~(size_t)255;
    return p;
}

static inline void gemm(hipStream_t s, const float* A, const float* B, float* C,
                        const float* bias, int M, int N, int K,
                        int lda, int ldb, int ldc, int accum,
                        int batch = 1, long sA = 0, long sB = 0, long sC = 0)
{
    dim3 g((unsigned)((N + TN - 1) / TN), (unsigned)((M + TM - 1) / TM), (unsigned)batch);
    gemm_bt_kernel<<<g, 128, 0, s>>>(A, B, C, bias, M, N, K, lda, ldb, ldc,
                                     sA, sB, sC, accum);
}

static inline unsigned blocksFor(long n) { return (unsigned)((n + 255) / 256); }

// 2-layer bidirectional LSTM with length masking.  x:[Nb,T,Din] -> out:[Nb,T,512]
// params: 16 pointers: {l0f, l0b, l1f, l1b} x {w_ih, w_hh, b_ih, b_hh}
static void run_bilstm(hipStream_t s, const float* x, const int* lens,
                       int Nb, int T, int Din,
                       const float* const* params, float* out, Bump tmp /*by value*/)
{
    const int Dmax = (Din > NH) ? Din : NH;
    float* currev = balloc(tmp, (size_t)Nb * T * Dmax);
    float* o0     = balloc(tmp, (size_t)Nb * T * NH);
    float* yb     = balloc(tmp, (size_t)Nb * T * HDIM);
    float* g      = balloc(tmp, (size_t)Nb * G4);
    float* hc     = balloc(tmp, (size_t)Nb * HDIM * 2);
    float* h = hc;
    float* c = hc + (size_t)Nb * HDIM;

    for (int l = 0; l < 2; ++l) {
        const int D = (l == 0) ? Din : NH;
        const float* in = (l == 0) ? x : o0;
        float* ob       = (l == 1) ? out : o0;
        const float* const* pf = params + l * 8;
        const float* const* pb = params + l * 8 + 4;
        const long total = (long)Nb * T * D;

        // reversed-by-length copy of this layer's input
        seq_copy_kernel<<<blocksFor(total), 256, 0, s>>>(
            currev, in, lens, Nb, T, D, (long)T * D, D, (long)T * D, D, 1);

        // forward direction -> writes ob[..., 0:256] directly
        fill_kernel<<<blocksFor((long)Nb * HDIM * 2), 256, 0, s>>>(hc, (long)Nb * HDIM * 2, 0.f);
        for (int t = 0; t < T; ++t) {
            gemm(s, in + (size_t)t * D, pf[0], g, nullptr, Nb, G4, D, T * D, D, G4, 0);
            gemm(s, h,                  pf[1], g, nullptr, Nb, G4, HDIM, HDIM, HDIM, G4, 1);
            lstm_gate_kernel<<<blocksFor((long)Nb * HDIM), 256, 0, s>>>(
                g, pf[2], pf[3], h, c, lens, t, Nb, ob, (long)T * NH, NH);
        }

        // backward direction over reversed input -> compact yb, then un-reverse
        fill_kernel<<<blocksFor((long)Nb * HDIM * 2), 256, 0, s>>>(hc, (long)Nb * HDIM * 2, 0.f);
        for (int t = 0; t < T; ++t) {
            gemm(s, currev + (size_t)t * D, pb[0], g, nullptr, Nb, G4, D, T * D, D, G4, 0);
            gemm(s, h,                      pb[1], g, nullptr, Nb, G4, HDIM, HDIM, HDIM, G4, 1);
            lstm_gate_kernel<<<blocksFor((long)Nb * HDIM), 256, 0, s>>>(
                g, pb[2], pb[3], h, c, lens, t, Nb, yb, (long)T * HDIM, HDIM);
        }
        seq_copy_kernel<<<blocksFor((long)Nb * T * HDIM), 256, 0, s>>>(
            ob + HDIM, yb, lens, Nb, T, HDIM,
            (long)T * HDIM, HDIM, (long)T * NH, NH, 1);
    }
}

extern "C" void kernel_launch(void* const* d_in, const int* in_sizes, int n_in,
                              void* d_out, int out_size, void* d_ws, size_t ws_size,
                              hipStream_t stream)
{
    (void)in_sizes; (void)n_in; (void)out_size; (void)ws_size;  // needs ~1.3 GB ws

    const float* q_emb    = (const float*)d_in[0];
    const int*   q_len    = (const int*)  d_in[1];
    const float* col_emb  = (const float*)d_in[2];
    const int*   col_nlen = (const int*)  d_in[3];
    const float* tab_emb  = (const float*)d_in[5];
    const int*   tab_nlen = (const int*)  d_in[6];
    const int*   col2tab  = (const int*)  d_in[8];
    const float* qp[16]; const float* cp[16];
    for (int i = 0; i < 16; ++i) qp[i] = (const float*)d_in[9 + i];
    for (int i = 0; i < 16; ++i) cp[i] = (const float*)d_in[25 + i];
    const float* att_w = (const float*)d_in[41];
    const float* att_b = (const float*)d_in[42];

    float* col_att = (float*)d_out;                         // [B, Tq, MC]
    float* tab_att = col_att + (size_t)BATCH * TQ * MC_;    // [B, Tq, MT]

    Bump ws{(char*)d_ws, 0};
    float* q_enc   = balloc(ws, (size_t)BATCH * TQ * NH);   // [B,Tq,512]
    float* col_enc = balloc(ws, (size_t)BATCH * MC_ * NH);  // [B,MC,512]
    float* tab_enc = balloc(ws, (size_t)BATCH * MT_ * NH);  // [B,MT,512]
    float* atted   = balloc(ws, (size_t)BATCH * TQ * NH);   // [B,Tq,512]
    const size_t mark = ws.off;

    // 1) question encoder
    run_bilstm(stream, q_emb, q_len, BATCH, TQ, NWORD, qp, q_enc, ws);

    // 2) column-name encoder + last-step extraction
    {
        Bump t = ws; t.off = mark;
        float* hid = balloc(t, (size_t)BATCH * MC_ * LC_ * NH);
        run_bilstm(stream, col_emb, col_nlen, BATCH * MC_, LC_, NWORD, cp, hid, t);
        take_last_kernel<<<blocksFor((long)BATCH * MC_ * NH), 256, 0, stream>>>(
            col_enc, hid, col_nlen, BATCH * MC_, NH, (long)LC_ * NH, NH);
    }

    // 3) table-name encoder + last-step extraction
    {
        Bump t = ws; t.off = mark;
        float* hid = balloc(t, (size_t)BATCH * MT_ * LT_ * NH);
        run_bilstm(stream, tab_emb, tab_nlen, BATCH * MT_, LT_, NWORD, cp, hid, t);
        take_last_kernel<<<blocksFor((long)BATCH * MT_ * NH), 256, 0, stream>>>(
            tab_enc, hid, tab_nlen, BATCH * MT_, NH, (long)LT_ * NH, NH);
    }

    // (col_len/tab_len are full MC/MT in this setup, so the zero-pad mask is a no-op)

    // 4) col_enc[b,c] += tab_enc[b, col2tab[b,c]]
    add_parent_kernel<<<blocksFor((long)BATCH * MC_ * NH), 256, 0, stream>>>(
        col_enc, tab_enc, col2tab, NH);

    // 5) atted = q_enc @ att_w^T + att_b     (M=32768, N=K=512)
    gemm(stream, q_enc, att_w, atted, att_b, BATCH * TQ, NH, NH, NH, NH, NH, 0);

    // 6) batched scores: col_att[b] = atted[b] @ col_enc[b]^T ; same for tables
    gemm(stream, atted, col_enc, col_att, nullptr, TQ, MC_, NH, NH, NH, MC_, 0,
         BATCH, (long)TQ * NH, (long)MC_ * NH, (long)TQ * MC_);
    gemm(stream, atted, tab_enc, tab_att, nullptr, TQ, MT_, NH, NH, NH, MT_, 0,
         BATCH, (long)TQ * NH, (long)MT_ * NH, (long)TQ * MT_);
}
